// NSF_CL_22376779612817
// MI455X (gfx1250) — matrix operand order repaired and checked by reading, not compiled, not run on hardware
//
#include <hip/hip_runtime.h>

// ---------------------------------------------------------------------------
// Types for CDNA5 WMMA (wave32): 16x16x32 bf16 -> f32, plus TDM descriptors
// ---------------------------------------------------------------------------
typedef __attribute__((ext_vector_type(16))) __bf16         bf16x16;
typedef __attribute__((ext_vector_type(8)))  float          f32x8;
typedef __attribute__((ext_vector_type(16))) unsigned short u16x16;
typedef __attribute__((ext_vector_type(8)))  unsigned short u16x8;
typedef __attribute__((ext_vector_type(4)))  unsigned int   u32x4;
typedef __attribute__((ext_vector_type(8)))  int            i32x8;
typedef __attribute__((ext_vector_type(4)))  int            i32x4;

__device__ __forceinline__ unsigned short f2bf(float f) {
    unsigned int u = __builtin_bit_cast(unsigned int, f);
    u += 0x7FFFu + ((u >> 16) & 1u);   // round-to-nearest-even
    return (unsigned short)(u >> 16);
}

__device__ __forceinline__ bf16x16 make_frag(u16x8 lo, u16x8 hi) {
    u16x16 v;
#pragma unroll
    for (int i = 0; i < 8; ++i) { v[i] = lo[i]; v[i + 8] = hi[i]; }
    return __builtin_bit_cast(bf16x16, v);
}

__device__ __forceinline__ f32x8 wmma_bf16(bf16x16 a, bf16x16 b, f32x8 c) {
    // v_wmma_f32_16x16x32_bf16  (neg_a, A, neg_b, B, c_mod, C, reuse_a, reuse_b)
    return __builtin_amdgcn_wmma_f32_16x16x32_bf16(false, a, false, b, (short)0, c, false, false);
}

// A-frag (16x32 bf16, row-major, row stride `ldr` elems):
//   lane m = lane&15 ; half = lane>>4 ; elems i -> K=kk+half*16+i
__device__ __forceinline__ bf16x16 load_a(const unsigned short* A, size_t row,
                                          int ldr, int kk, int half) {
    const unsigned short* ap = A + row * (size_t)ldr + kk + half * 16;
    return make_frag(*(const u16x8*)ap, *(const u16x8*)(ap + 8));
}
// B-frag (32x16 bf16 from transposed weights Wt[N x K]): lane n = lane&15,
//   elems i -> K = kk + half*16 + i  (single contiguous 32B run)
__device__ __forceinline__ bf16x16 load_b(const unsigned short* Wt, size_t col,
                                          int K, int kk, int half) {
    const unsigned short* bp = Wt + col * (size_t)K + kk + half * 16;
    return __builtin_bit_cast(bf16x16, *(const u16x16*)bp);
}

// ---------------------------------------------------------------------------
// TDM: DMA a 16x512 bf16 tile (row stride 512 elems) from global into LDS,
// with 16B padding appended per 1024B row (pad_interval=256dw, pad_amount=4dw)
// so the LDS row stride is 1040B -> conflict-free ds_load_b128 A-fragments.
// ---------------------------------------------------------------------------
__device__ __forceinline__ void tdm_load_tile_16x512(unsigned lds_addr,
                                                     const unsigned short* gptr) {
    unsigned long long ga = (unsigned long long)(uintptr_t)gptr;
    u32x4 g0;
    g0[0] = 1u;                                          // count=1 (user descriptor)
    g0[1] = lds_addr;                                    // LDS byte address
    g0[2] = (unsigned)(ga & 0xFFFFFFFFull);              // global_addr[31:0]
    g0[3] = (unsigned)((ga >> 32) & 0x01FFFFFFull) | 0x80000000u;  // addr[56:32] | type=2
    i32x8 g1 = {};
    g1[0] = (int)0x07D10000u;   // data_size=2B | pad_enable | pad_interval=7 | pad_amount=3
    g1[1] = (int)(512u << 16);  // tensor_dim0 = 512  (bits 63:48 = low16)
    g1[2] = (int)(16u << 16);   // tensor_dim0 hi=0 | tensor_dim1 = 16
    g1[3] = (int)(512u << 16);  // tensor_dim1 hi=0 | tile_dim0 = 512
    g1[4] = 16;                 // tile_dim1 = 16, tile_dim2 = 0
    g1[5] = 512;                // tensor_dim0_stride = 512
    i32x4 gz = {};
#if defined(__clang_major__) && __clang_major__ >= 23
    i32x8 gz8 = {};
    __builtin_amdgcn_tensor_load_to_lds(g0, g1, gz, gz, gz8, 0);
#else
    __builtin_amdgcn_tensor_load_to_lds(g0, g1, gz, gz, 0);
#endif
}

// ---------------------------------------------------------------------------
// Rational-quadratic spline (K=8 bins, B_TAIL=1, MIN_*=1e-3), branch-free
// ---------------------------------------------------------------------------
__device__ __forceinline__ void rqs_eval(float x, const float* __restrict__ p,
                                         float& y, float& lad) {
    const float MW = 0.001f, MH = 0.001f, MD = 0.001f;
    float ew[8], eh[8];
    float mw = p[0], mh = p[8];
#pragma unroll
    for (int i = 1; i < 8; ++i) { mw = fmaxf(mw, p[i]); mh = fmaxf(mh, p[8 + i]); }
    float sw = 0.f, sh = 0.f;
#pragma unroll
    for (int i = 0; i < 8; ++i) {
        ew[i] = __expf(p[i] - mw);     sw += ew[i];
        eh[i] = __expf(p[8 + i] - mh); sh += eh[i];
    }
    float rw = (1.0f - 8.0f * MW) / sw, rh = (1.0f - 8.0f * MH) / sh;
    float cw[9], ch[9];
    cw[0] = -1.0f; ch[0] = -1.0f;
    float aw = 0.f, ah = 0.f;
#pragma unroll
    for (int i = 0; i < 8; ++i) {
        aw += MW + ew[i] * rw; cw[i + 1] = 2.0f * aw - 1.0f;
        ah += MH + eh[i] * rh; ch[i + 1] = 2.0f * ah - 1.0f;
    }
    cw[8] = 1.0f; ch[8] = 1.0f;
    float dv[9];
    dv[0] = 1.0f; dv[8] = 1.0f;                       // softplus(const)+MIN_D == 1
#pragma unroll
    for (int i = 0; i < 7; ++i) {
        float z = p[16 + i];                          // stable softplus
        dv[i + 1] = MD + fmaxf(z, 0.0f) + log1pf(__expf(-fabsf(z)));
    }
    int idx = 0;
#pragma unroll
    for (int i = 1; i < 8; ++i) idx += (x >= cw[i]) ? 1 : 0;

    float icw = cw[0], icw1 = cw[1], ich = ch[0], ich1 = ch[1], id0 = dv[0], id1 = dv[1];
#pragma unroll
    for (int i = 1; i < 8; ++i) {
        bool m = (idx == i);
        icw  = m ? cw[i]     : icw;
        icw1 = m ? cw[i + 1] : icw1;
        ich  = m ? ch[i]     : ich;
        ich1 = m ? ch[i + 1] : ich1;
        id0  = m ? dv[i]     : id0;
        id1  = m ? dv[i + 1] : id1;
    }
    float wid   = icw1 - icw;
    float hei   = ich1 - ich;
    float delta = hei / wid;
    float th    = (x - icw) / wid;
    float omt   = 1.0f - th;
    float t1    = th * omt;
    float numer = hei * (delta * th * th + id0 * t1);
    float denom = delta + (id0 + id1 - 2.0f * delta) * t1;
    float outv  = ich + numer / denom;
    float dnum  = delta * delta * (id1 * th * th + 2.0f * delta * t1 + id0 * omt * omt);
    float l     = __logf(dnum) - 2.0f * __logf(denom);
    bool inside = (x >= -1.0f) && (x <= 1.0f);
    y   = inside ? outv : x;
    lad = inside ? l : 0.0f;
}

// ---------------------------------------------------------------------------
// Small converters: f32 weight [R x C] -> bf16 transposed [C x R]; f32 slice -> bf16
// ---------------------------------------------------------------------------
__global__ void k_transpose_bf16(const float* __restrict__ in,
                                 unsigned short* __restrict__ out, int R, int C) {
    int idx = blockIdx.x * 256 + threadIdx.x;
    if (idx < R * C) {
        int c = idx / R, r = idx % R;                 // out[c*R + r] = in[r*C + c]
        out[idx] = f2bf(in[(size_t)r * C + c]);
    }
}

__global__ void k_slice_bf16(const float* __restrict__ x,
                             unsigned short* __restrict__ out,
                             int M, int cols, int ld, int cb) {
    int idx = blockIdx.x * 256 + threadIdx.x;
    if (idx < M * cols) {
        int r = idx / cols, c = idx % cols;
        out[idx] = f2bf(x[(size_t)r * ld + cb + c]);
    }
}

// ---------------------------------------------------------------------------
// GEMM + SiLU: H[MxN] = silu(A[MxK] * Wt^T + b)   (A,H bf16, Wt [NxK] bf16)
// 256 threads = 8 waves; WG tile 128x64; wave tile 32x32 (2x2 WMMA)
// ---------------------------------------------------------------------------
__global__ __launch_bounds__(256) void k_gemm_silu(
    const unsigned short* __restrict__ A, const unsigned short* __restrict__ Wt,
    const float* __restrict__ bias, unsigned short* __restrict__ H,
    int M, int N, int K) {
    (void)M;
    const int tid  = threadIdx.x;
    const int wave = tid >> 5, lane = tid & 31;
    const int half = lane >> 4, l16 = lane & 15;
    const int tileM = blockIdx.x * 128 + (wave & 3) * 32;
    const int tileN = blockIdx.y * 64 + (wave >> 2) * 32;

    f32x8 acc[2][2] = {};
    for (int kk = 0; kk < K; kk += 32) {
        if (kk + 32 < K)  // stream next weight K-slab (global_prefetch_b8)
            __builtin_prefetch(Wt + (size_t)(tileN + l16) * K + kk + 32, 0, 0);
        bf16x16 a[2], b[2];
#pragma unroll
        for (int mi = 0; mi < 2; ++mi)
            a[mi] = load_a(A, (size_t)(tileM + mi * 16 + l16), K, kk, half);
#pragma unroll
        for (int ni = 0; ni < 2; ++ni)
            b[ni] = load_b(Wt, (size_t)(tileN + ni * 16 + l16), K, kk, half);
#pragma unroll
        for (int mi = 0; mi < 2; ++mi)
#pragma unroll
            for (int ni = 0; ni < 2; ++ni)
                acc[mi][ni] = wmma_bf16(a[mi], b[ni], acc[mi][ni]);
    }
    // epilogue: bias + SiLU, store bf16
#pragma unroll
    for (int mi = 0; mi < 2; ++mi) {
#pragma unroll
        for (int ni = 0; ni < 2; ++ni) {
            int col  = tileN + ni * 16 + l16;
            float bv = bias[col];
#pragma unroll
            for (int r = 0; r < 8; ++r) {
                int row = tileM + mi * 16 + half * 8 + r;  // C layout: M = r + 8*half
                float s = acc[mi][ni][r] + bv;
                float o = s / (1.0f + __expf(-s));
                H[(size_t)row * N + col] = f2bf(o);
            }
        }
    }
}

// ---------------------------------------------------------------------------
// GEMM3 fused with RQS spline.  WG = 16 rows x full N=1472 (92 N-tiles of 16),
// 8 waves x 12 tiles each.  The 16x512 bf16 A-tile is DMA'd once per WG into
// LDS by the Tensor Data Mover (with descriptor padding: LDS row stride 1040B
// -> bank-conflict-free A-fragments), synchronized via TENSORcnt.  Params are
// staged f32 in LDS (16x1472 = 94KB), then 16x64 spline evals + LDS log-det
// reduction.  Avoids materializing the 193MB parameter tensor in HBM.
// ---------------------------------------------------------------------------
#define NTPW 12
#define ALDS 520   // padded LDS row stride in ushorts (1040 B)
__global__ __launch_bounds__(256) void k_gemm_spline(
    const unsigned short* __restrict__ A,    // [M x 512] bf16
    const unsigned short* __restrict__ Wt,   // [1472 x 512] bf16 (transposed w3)
    const float* __restrict__ bias,          // [1472]
    const float* __restrict__ xsrc,          // x (+col offset), row stride 128
    float* __restrict__ yout,                // d_out y, row stride 128
    int outbase,
    float* __restrict__ ldout,               // d_out log_det [M]
    unsigned short* __restrict__ nextbf,     // bf16 [M x 64] for next FCNN, or null
    int accumulate) {
    __shared__ unsigned short Astage[16 * ALDS];   // 16.6 KB TDM landing zone
    extern __shared__ float smem[];
    float* P  = smem;               // 16 x 1472 spline params
    float* LR = smem + 16 * 1472;   // 16 x 16 log-det partials

    const int tid  = threadIdx.x;
    const int wave = tid >> 5, lane = tid & 31;
    const int half = lane >> 4, l16 = lane & 15;
    const int tileM = blockIdx.x * 16;
    const int K = 512;

    // One wave issues the tensor DMA for the whole workgroup (EXEC-independent,
    // tracked by TENSORcnt); everyone meets at the barrier.
    if (wave == 0) {
        tdm_load_tile_16x512((unsigned)(uintptr_t)(void*)Astage,
                             A + (size_t)tileM * K);
        __builtin_amdgcn_s_wait_tensorcnt(0);
    }
    __syncthreads();

    f32x8 acc[NTPW] = {};
    for (int kk = 0; kk < K; kk += 32) {
        bf16x16 a = load_a(Astage, (size_t)l16, ALDS, kk, half);   // ds_load_b128
        if (kk + 32 < K)
            __builtin_prefetch(Wt + (size_t)(wave * NTPW * 16 + l16) * K + kk + 32, 0, 0);
#pragma unroll
        for (int j = 0; j < NTPW; ++j) {
            int t = wave * NTPW + j;                 // wave-uniform predicate
            if (t < 92) {
                bf16x16 b = load_b(Wt, (size_t)(t * 16 + l16), K, kk, half);
                acc[j] = wmma_bf16(a, b, acc[j]);
            }
        }
    }
    // accumulators + bias -> LDS
#pragma unroll
    for (int j = 0; j < NTPW; ++j) {
        int t = wave * NTPW + j;
        if (t < 92) {
            int col  = t * 16 + l16;
            float bv = bias[col];
#pragma unroll
            for (int r = 0; r < 8; ++r)
                P[(half * 8 + r) * 1472 + col] = acc[j][r] + bv;
        }
    }
    __syncthreads();

    // spline phase: thread -> (row = tid&15, dims = tid>>4 + 16*i)
    const int row   = tid & 15;
    const int dbase = tid >> 4;
    const int rowg  = tileM + row;
    float lsum = 0.0f;
#pragma unroll
    for (int i = 0; i < 4; ++i) {
        int dim = dbase + 16 * i;
        const float* p = P + row * 1472 + dim * 23;  // [W(8) H(8) D(7)]
        float xv = xsrc[(size_t)rowg * 128 + dim];
        float yv, lad;
        rqs_eval(xv, p, yv, lad);
        yout[(size_t)rowg * 128 + outbase + dim] = yv;
        if (nextbf) nextbf[(size_t)rowg * 64 + dim] = f2bf(yv);
        lsum += lad;
    }
    LR[row * 16 + dbase] = lsum;
    __syncthreads();
    if (tid < 16) {
        float s = 0.0f;
#pragma unroll
        for (int i = 0; i < 16; ++i) s += LR[tid * 16 + i];
        int rg = tileM + tid;
        ldout[rg] = accumulate ? (ldout[rg] + s) : s;
    }
}

// ---------------------------------------------------------------------------
// Host launch: 6-stage pipeline (FCNN1 -> spline(upper) -> FCNN2 -> spline(lower))
// ---------------------------------------------------------------------------
extern "C" void kernel_launch(void* const* d_in, const int* in_sizes, int n_in,
                              void* d_out, int out_size, void* d_ws, size_t ws_size,
                              hipStream_t stream) {
    (void)in_sizes; (void)n_in; (void)out_size; (void)ws_size;
    const float* x    = (const float*)d_in[0];
    const float* f1w1 = (const float*)d_in[1];  const float* f1b1 = (const float*)d_in[2];
    const float* f1w2 = (const float*)d_in[3];  const float* f1b2 = (const float*)d_in[4];
    const float* f1w3 = (const float*)d_in[5];  const float* f1b3 = (const float*)d_in[6];
    const float* f2w1 = (const float*)d_in[7];  const float* f2b1 = (const float*)d_in[8];
    const float* f2w2 = (const float*)d_in[9];  const float* f2b2 = (const float*)d_in[10];
    const float* f2w3 = (const float*)d_in[11]; const float* f2b3 = (const float*)d_in[12];

    const int M = 32768, HD = 512, OD = 1472, HF = 64;
    unsigned short* ws = (unsigned short*)d_ws;
    // workspace layout (bf16 elements); total ~76 MB
    const size_t W1T1 = 0;
    const size_t W2T1 = W1T1 + (size_t)HD * HF;
    const size_t W3T1 = W2T1 + (size_t)HD * HD;
    const size_t W1T2 = W3T1 + (size_t)OD * HD;
    const size_t W2T2 = W1T2 + (size_t)HD * HF;
    const size_t W3T2 = W2T2 + (size_t)HD * HD;
    const size_t XLOW = W3T2 + (size_t)OD * HD;
    const size_t XUP  = XLOW + (size_t)M * HF;
    const size_t H1   = XUP + (size_t)M * HF;
    const size_t H2   = H1 + (size_t)M * HD;

    float* y  = (float*)d_out;
    float* ld = y + (size_t)M * 128;

    auto cdiv = [](int a, int b) { return (a + b - 1) / b; };
    k_transpose_bf16<<<cdiv(HF * HD, 256), 256, 0, stream>>>(f1w1, ws + W1T1, HF, HD);
    k_transpose_bf16<<<cdiv(HD * HD, 256), 256, 0, stream>>>(f1w2, ws + W2T1, HD, HD);
    k_transpose_bf16<<<cdiv(HD * OD, 256), 256, 0, stream>>>(f1w3, ws + W3T1, HD, OD);
    k_transpose_bf16<<<cdiv(HF * HD, 256), 256, 0, stream>>>(f2w1, ws + W1T2, HF, HD);
    k_transpose_bf16<<<cdiv(HD * HD, 256), 256, 0, stream>>>(f2w2, ws + W2T2, HD, HD);
    k_transpose_bf16<<<cdiv(HD * OD, 256), 256, 0, stream>>>(f2w3, ws + W3T2, HD, OD);
    k_slice_bf16<<<cdiv(M * HF, 256), 256, 0, stream>>>(x, ws + XLOW, M, HF, 128, 0);

    dim3 g1(M / 128, HD / 64);
    size_t smem = (size_t)(16 * 1472 + 16 * 16) * sizeof(float);

    // FCNN1 on lower -> spline transforms upper (cols 64..127), ld1
    k_gemm_silu<<<g1, 256, 0, stream>>>(ws + XLOW, ws + W1T1, f1b1, ws + H1, M, HD, HF);
    k_gemm_silu<<<g1, 256, 0, stream>>>(ws + H1,  ws + W2T1, f1b2, ws + H2, M, HD, HD);
    k_gemm_spline<<<M / 16, 256, smem, stream>>>(ws + H2, ws + W3T1, f1b3,
                                                 x + 64, y, 64, ld, ws + XUP, 0);
    // FCNN2 on transformed upper -> spline transforms lower (cols 0..63), ld += ld2
    k_gemm_silu<<<g1, 256, 0, stream>>>(ws + XUP, ws + W1T2, f2b1, ws + H1, M, HD, HF);
    k_gemm_silu<<<g1, 256, 0, stream>>>(ws + H1,  ws + W2T2, f2b2, ws + H2, M, HD, HD);
    k_gemm_spline<<<M / 16, 256, smem, stream>>>(ws + H2, ws + W3T2, f2b3,
                                                 x, y, 0, ld, (unsigned short*)nullptr, 1);
}
